// NetworkSchemaEncoder_2044404433796
// MI455X (gfx1250) — compile-verified, matrix operand order checked by reading, not run on hardware
//
#include <hip/hip_runtime.h>
#include <hip/hip_bf16.h>
#include <math.h>

typedef __attribute__((ext_vector_type(16))) _Float16 v16h;
typedef __attribute__((ext_vector_type(8)))  float    v8f;

#define D 64

// ---------------------------------------------------------------------------
// Utility kernels
// ---------------------------------------------------------------------------
__global__ void fill_f32(float* __restrict__ p, float v, size_t n) {
    size_t i = (size_t)blockIdx.x * blockDim.x + threadIdx.x;
    if (i < n) p[i] = v;
}

__global__ void f32_to_f16_k(const float* __restrict__ w, _Float16* __restrict__ o, int n) {
    int i = blockIdx.x * blockDim.x + threadIdx.x;
    if (i < n) o[i] = (_Float16)w[i];
}

// ---------------------------------------------------------------------------
// Per-node attention logits: out[n] = feat[n,:] . attn  (one wave32 per node)
// ---------------------------------------------------------------------------
__global__ void node_logits(const float* __restrict__ feat, const float* __restrict__ attn,
                            int n, float* __restrict__ out) {
    int wave = (int)((blockIdx.x * (size_t)blockDim.x + threadIdx.x) >> 5);
    int lane = threadIdx.x & 31;
    if (wave >= n) return;
    const float* row = feat + (size_t)wave * D;
    float s = row[lane] * attn[lane] + row[lane + 32] * attn[lane + 32];
    #pragma unroll
    for (int off = 16; off > 0; off >>= 1) s += __shfl_xor(s, off, 32);
    if (lane == 0) out[wave] = s;
}

// ---------------------------------------------------------------------------
// Edge pass 1: e = leaky_relu(el[src] + er[dst]); segment max into emax[dst]
// ---------------------------------------------------------------------------
__global__ void edge_logits_max(const float* __restrict__ el, const float* __restrict__ er,
                                const int* __restrict__ src, const int* __restrict__ dst,
                                float* __restrict__ e_out, float* __restrict__ emax, int E) {
    int i = blockIdx.x * blockDim.x + threadIdx.x;
    if (i >= E) return;
    float e = el[src[i]] + er[dst[i]];
    e = (e > 0.f) ? e : 0.01f * e;
    e_out[i] = e;
    atomicMax(&emax[dst[i]], e);
}

// ---------------------------------------------------------------------------
// Edge pass 2: ee = exp(e - emax[dst]); segment sum into denom[dst]
// ---------------------------------------------------------------------------
__global__ void edge_exp_sum(const int* __restrict__ dst, const float* __restrict__ emax,
                             float* __restrict__ e_inout, float* __restrict__ denom, int E) {
    int i = blockIdx.x * blockDim.x + threadIdx.x;
    if (i >= E) return;
    int d = dst[i];
    float ee = __expf(e_inout[i] - emax[d]);
    e_inout[i] = ee;
    atomicAdd(&denom[d], ee);
}

// ---------------------------------------------------------------------------
// Edge pass 3: hbuf[dst,:] += (ee/denom[dst]) * feat_src[src,:]  (64 lanes/edge)
// ---------------------------------------------------------------------------
__global__ void edge_aggregate(const float* __restrict__ feat_src,
                               const int* __restrict__ src, const int* __restrict__ dst,
                               const float* __restrict__ ee, const float* __restrict__ denom,
                               float* __restrict__ hbuf, int E) {
    size_t gid = (size_t)blockIdx.x * blockDim.x + threadIdx.x;
    int e = (int)(gid >> 6);
    int k = (int)(gid & 63);
    if (e >= E) return;
    int d = dst[e];
    float a = ee[e] / denom[d];
    atomicAdd(&hbuf[(size_t)d * D + k], a * feat_src[(size_t)src[e] * D + k]);
}

// ---------------------------------------------------------------------------
// In-place ELU
// ---------------------------------------------------------------------------
__global__ void elu_k(float* __restrict__ p, size_t n) {
    size_t i = (size_t)blockIdx.x * blockDim.x + threadIdx.x;
    if (i >= n) return;
    float x = p[i];
    p[i] = (x > 0.f) ? x : (__expf(x) - 1.f);
}

// ---------------------------------------------------------------------------
// Semantic attention stage 1 (WMMA).
// Per-metapath helper: one wave computes a 16x64 tile of tanh(h @ fc_w^T + b),
// column-sums it and atomically accumulates into sacc[m*64 + j].
// A/B per-lane layout per ISA 7.12.2 (16-bit operands):
//   k(h) = (h/8)*16 + hi*8 + (h%8) + kh*32  -> two contiguous 8-float runs.
// ---------------------------------------------------------------------------
__device__ __forceinline__ void semantic_tile(const float* __restrict__ hrow,
                                              const _Float16* lw,       // LDS 64x64 f16
                                              const float* __restrict__ fcb,
                                              float* __restrict__ sacc,
                                              int m, int hi, int col) {
    v8f c[4];
    #pragma unroll
    for (int t = 0; t < 4; ++t)
        #pragma unroll
        for (int r = 0; r < 8; ++r) c[t][r] = 0.f;

    #pragma unroll
    for (int kh = 0; kh < 2; ++kh) {
        // A fragment: two contiguous 8-float runs -> explicit float4 loads.
        const int base = hi * 8 + kh * 32;
        const float4* p0 = (const float4*)(hrow + base);        // k = base..base+7
        const float4* p1 = (const float4*)(hrow + base + 16);   // k = base+16..base+23
        float4 r0 = p0[0], r1 = p0[1], r2 = p1[0], r3 = p1[1];
        v16h a;
        a[0]  = (_Float16)r0.x; a[1]  = (_Float16)r0.y; a[2]  = (_Float16)r0.z; a[3]  = (_Float16)r0.w;
        a[4]  = (_Float16)r1.x; a[5]  = (_Float16)r1.y; a[6]  = (_Float16)r1.z; a[7]  = (_Float16)r1.w;
        a[8]  = (_Float16)r2.x; a[9]  = (_Float16)r2.y; a[10] = (_Float16)r2.z; a[11] = (_Float16)r2.w;
        a[12] = (_Float16)r3.x; a[13] = (_Float16)r3.y; a[14] = (_Float16)r3.z; a[15] = (_Float16)r3.w;

        #pragma unroll
        for (int t = 0; t < 4; ++t) {
            int j = t * 16 + col;
            v16h b;
            #pragma unroll
            for (int h = 0; h < 16; ++h) {
                int k = ((h >> 3) << 4) + hi * 8 + (h & 7) + kh * 32;
                b[h] = lw[j * D + k];
            }
            c[t] = __builtin_amdgcn_wmma_f32_16x16x32_f16(
                false, a, false, b, (short)0, c[t], false, false);
        }
    }
    // C layout: VGPR r holds row M = r + hi*8, column N = col.
    #pragma unroll
    for (int t = 0; t < 4; ++t) {
        int j = t * 16 + col;
        float bj = fcb[j];
        float s = 0.f;
        #pragma unroll
        for (int r = 0; r < 8; ++r) s += tanhf(c[t][r] + bj);
        s += __shfl_xor(s, 16, 32);    // rows 0-7 + rows 8-15
        if (hi == 0) atomicAdd(&sacc[m * D + j], s);
    }
}

__global__ void __launch_bounds__(256)
semantic_stage1(const float* __restrict__ h_ap, const float* __restrict__ h_sp,
                const _Float16* __restrict__ fcw16, const float* __restrict__ fcb,
                float* __restrict__ sacc, int n_tiles, int n_paper) {
    __shared__ _Float16 lw[D * D];
    for (int i = threadIdx.x; i < D * D; i += 256) lw[i] = fcw16[i];
    __syncthreads();

    int wave = threadIdx.x >> 5;
    int lane = threadIdx.x & 31;
    int tile = blockIdx.x * 8 + wave;
    if (tile >= n_tiles) return;

    int hi  = lane >> 4;
    int col = lane & 15;
    int row = tile * 16 + col;
    if (row >= n_paper) row = n_paper - 1;   // n_paper is a multiple of 16 here
    size_t roff = (size_t)row * D;

    // Manual metapath unroll: distinct uniform global pointers so the backend
    // emits global_load_b128 (flat loads would also burn DScnt on CDNA5).
    semantic_tile(h_ap + roff, lw, fcb, sacc, 0, hi, col);
    semantic_tile(h_sp + roff, lw, fcb, sacc, 1, hi, col);
}

// ---------------------------------------------------------------------------
// Semantic attention stage 2: w[m] = (sacc[m]/N) . sem_attn; beta = softmax(w)
// ---------------------------------------------------------------------------
__global__ void semantic_stage2(const float* __restrict__ sacc, const float* __restrict__ sem,
                                float* __restrict__ wbeta, float inv_n) {
    if (blockIdx.x == 0 && threadIdx.x == 0) {
        float w0 = 0.f, w1 = 0.f;
        for (int j = 0; j < D; ++j) {
            w0 += sacc[j] * inv_n * sem[j];
            w1 += sacc[D + j] * inv_n * sem[j];
        }
        float mx = fmaxf(w0, w1);
        float e0 = __expf(w0 - mx), e1 = __expf(w1 - mx);
        float dsum = e0 + e1;
        wbeta[0] = e0 / dsum;
        wbeta[1] = e1 / dsum;
    }
}

// ---------------------------------------------------------------------------
// Final combine: out = beta0*h_ap + beta1*h_sp
// ---------------------------------------------------------------------------
__global__ void combine_k(const float* __restrict__ h_ap, const float* __restrict__ h_sp,
                          const float* __restrict__ wbeta, float* __restrict__ out, size_t n) {
    size_t i = (size_t)blockIdx.x * blockDim.x + threadIdx.x;
    if (i >= n) return;
    out[i] = wbeta[0] * h_ap[i] + wbeta[1] * h_sp[i];
}

// ---------------------------------------------------------------------------
// Launch
// ---------------------------------------------------------------------------
extern "C" void kernel_launch(void* const* d_in, const int* in_sizes, int n_in,
                              void* d_out, int out_size, void* d_ws, size_t ws_size,
                              hipStream_t stream) {
    const float* feat_author  = (const float*)d_in[0];
    const float* feat_subject = (const float*)d_in[1];
    const float* feat_paper   = (const float*)d_in[2];
    const int*   src_ap       = (const int*)d_in[3];
    const int*   dst_ap       = (const int*)d_in[4];
    const int*   src_sp       = (const int*)d_in[5];
    const int*   dst_sp       = (const int*)d_in[6];
    const float* attn_l_ap    = (const float*)d_in[7];
    const float* attn_r_ap    = (const float*)d_in[8];
    const float* attn_l_sp    = (const float*)d_in[9];
    const float* attn_r_sp    = (const float*)d_in[10];
    const float* fc_w         = (const float*)d_in[11];
    const float* fc_b         = (const float*)d_in[12];
    const float* sem_attn     = (const float*)d_in[13];
    float*       out          = (float*)d_out;

    const int n_author  = in_sizes[0] / D;
    const int n_subject = in_sizes[1] / D;
    const int n_paper   = in_sizes[2] / D;
    const int E_ap      = in_sizes[3];
    const int E_sp      = in_sizes[5];

    // ---- workspace carve-out (float units, 64-element aligned) ----
    float* base = (float*)d_ws;
    size_t off = 0;
    auto alloc = [&](size_t nf) { size_t o = off; off += (nf + 63) & ~(size_t)63; return base + o; };

    float* h_ap     = alloc((size_t)n_paper * D);
    float* h_sp     = alloc((size_t)n_paper * D);
    float* denom_ap = alloc(n_paper);
    float* denom_sp = alloc(n_paper);
    float* sacc     = alloc(2 * D);
    float* wbeta    = alloc(8);
    size_t zero_count = (size_t)((wbeta + 64) - h_ap);   // contiguous zero region

    float* emax_ap  = alloc(n_paper);
    float* emax_sp  = alloc(n_paper);
    size_t ninf_count = (size_t)((emax_sp + ((n_paper + 63) & ~63)) - emax_ap);

    float* el_ap    = alloc(n_author);
    float* er_ap    = alloc(n_paper);
    float* el_sp    = alloc(n_subject);
    float* er_sp    = alloc(n_paper);
    float* ee_ap    = alloc(E_ap);
    float* ee_sp    = alloc(E_sp);
    _Float16* fcw16 = (_Float16*)alloc(D * D / 2 + 64);

    const int B = 256;
    auto blk = [](size_t n, int b) { return (unsigned)((n + b - 1) / b); };

    // ---- init accumulators (every call: graph-replay safe) ----
    fill_f32<<<blk(zero_count, B), B, 0, stream>>>(h_ap, 0.f, zero_count);
    fill_f32<<<blk(ninf_count, B), B, 0, stream>>>(emax_ap, -INFINITY, ninf_count);
    f32_to_f16_k<<<blk(D * D, B), B, 0, stream>>>(fc_w, fcw16, D * D);

    // ---- per-node attention logits (wave per node) ----
    node_logits<<<blk((size_t)n_author * 32, B), B, 0, stream>>>(feat_author, attn_l_ap, n_author, el_ap);
    node_logits<<<blk((size_t)n_paper  * 32, B), B, 0, stream>>>(feat_paper,  attn_r_ap, n_paper,  er_ap);
    node_logits<<<blk((size_t)n_subject* 32, B), B, 0, stream>>>(feat_subject,attn_l_sp, n_subject,el_sp);
    node_logits<<<blk((size_t)n_paper  * 32, B), B, 0, stream>>>(feat_paper,  attn_r_sp, n_paper,  er_sp);

    // ---- edge softmax + aggregation, metapath AP ----
    edge_logits_max<<<blk(E_ap, B), B, 0, stream>>>(el_ap, er_ap, src_ap, dst_ap, ee_ap, emax_ap, E_ap);
    edge_exp_sum  <<<blk(E_ap, B), B, 0, stream>>>(dst_ap, emax_ap, ee_ap, denom_ap, E_ap);
    edge_aggregate<<<blk((size_t)E_ap * D, B), B, 0, stream>>>(feat_author, src_ap, dst_ap, ee_ap, denom_ap, h_ap, E_ap);

    // ---- edge softmax + aggregation, metapath SP ----
    edge_logits_max<<<blk(E_sp, B), B, 0, stream>>>(el_sp, er_sp, src_sp, dst_sp, ee_sp, emax_sp, E_sp);
    edge_exp_sum  <<<blk(E_sp, B), B, 0, stream>>>(dst_sp, emax_sp, ee_sp, denom_sp, E_sp);
    edge_aggregate<<<blk((size_t)E_sp * D, B), B, 0, stream>>>(feat_subject, src_sp, dst_sp, ee_sp, denom_sp, h_sp, E_sp);

    // ---- ELU ----
    elu_k<<<blk((size_t)n_paper * D, B), B, 0, stream>>>(h_ap, (size_t)n_paper * D);
    elu_k<<<blk((size_t)n_paper * D, B), B, 0, stream>>>(h_sp, (size_t)n_paper * D);

    // ---- semantic attention (WMMA GEMM + softmax + combine) ----
    int n_tiles = (n_paper + 15) / 16;
    semantic_stage1<<<blk((size_t)n_tiles * 32, B), B, 0, stream>>>(h_ap, h_sp, fcw16, fc_b, sacc, n_tiles, n_paper);
    semantic_stage2<<<1, 32, 0, stream>>>(sacc, sem_attn, wbeta, 1.f / (float)n_paper);
    combine_k<<<blk((size_t)n_paper * D, B), B, 0, stream>>>(h_ap, h_sp, wbeta, out, (size_t)n_paper * D);
}
